// SNNRegressor_78211354460331
// MI455X (gfx1250) — matrix operand order repaired and checked by reading, not compiled, not run on hardware
//
#include <hip/hip_runtime.h>

typedef __attribute__((ext_vector_type(16))) _Float16 v16h;
typedef __attribute__((ext_vector_type(8)))  float    v8f;
typedef int vi4 __attribute__((vector_size(16)));

#define THR  1.0f
#define BETA 0.9f

#if __has_builtin(__builtin_amdgcn_global_load_async_to_lds_b32) && \
    __has_builtin(__builtin_amdgcn_global_load_async_to_lds_b128) && \
    __has_builtin(__builtin_amdgcn_s_wait_asynccnt)
#define USE_ASYNC 1
#else
#define USE_ASYNC 0
#endif

#define GAS __attribute__((address_space(1)))
#define LAS __attribute__((address_space(3)))

#if USE_ASYNC
__device__ __forceinline__ void async_g2l_b32(const void* g, void* l) {
  __builtin_amdgcn_global_load_async_to_lds_b32((GAS int*)g, (LAS int*)l, 0, 0);
}
__device__ __forceinline__ void async_g2l_b128(const void* g, void* l) {
  __builtin_amdgcn_global_load_async_to_lds_b128((GAS vi4*)g, (LAS vi4*)l, 0, 0);
}
#endif

// ---------------------------------------------------------------------------
// B-fragment layout (f16 B 32x16, wave32): lane L, element e: N=L%16, K=(L/16)*16+e.
// Fragments pre-packed as [frag][lane][16 halfs] -> one 32B contiguous load/lane.
// A-fragment (f16 A 16x32): lane L: M=L%16; element e: K=(L/16)*16+e.
// Weights are split-f16: hi = f16(w), lo = f16(w - hi); two chained WMMAs give
// ~f32-accurate products since the spike A-matrix is exactly {0,1}.
// ---------------------------------------------------------------------------

__device__ __forceinline__ v16h load_bfrag(const unsigned short* base, int lane) {
  union { v16h v; uint4 q[2]; } u;
  const uint4* p = reinterpret_cast<const uint4*>(base + lane * 16);
  u.q[0] = p[0];
  u.q[1] = p[1];
  return u.v;
}

__device__ __forceinline__ v16h load_afrag(const unsigned short* lds, int stride_h,
                                           int kbase, int lane) {
  const int M  = lane & 15;
  const int hi = (lane >> 4) & 1;
  union { v16h v; unsigned u32[8]; } r;
  const unsigned* p = reinterpret_cast<const unsigned*>(lds);
  const int rowb = (M * stride_h) >> 1;
  #pragma unroll
  for (int vv = 0; vv < 8; ++vv) {
    const int k = kbase + (hi << 4) + (vv << 1);
    r.u32[vv] = p[rowb + (k >> 1)];
  }
  return r.v;
}

__device__ __forceinline__ void lif_update_store(v8f cur, v8f& mem, float bias,
                                                 unsigned short* spkLds, int stride_h,
                                                 int nbase, int lane) {
  const int l16 = lane & 15;
  const int hi  = (lane >> 4) & 1;
  #pragma unroll
  for (int r = 0; r < 8; ++r) {
    float c  = cur[r] + bias;
    float mm = mem[r];
    const float reset = (mm > THR) ? THR : 0.0f;
    mm = BETA * mm + c - reset;
    mem[r] = mm;
    const int M = r + (hi << 3);
    spkLds[M * stride_h + nbase + l16] =
        (mm > THR) ? (unsigned short)0x3C00 : (unsigned short)0;
  }
}

// ---------------------------------------------------------------------------
// Setup kernels
// ---------------------------------------------------------------------------
__global__ void snn_probs(const float* __restrict__ x, float* __restrict__ pIn,
                          float* __restrict__ pTd, float* __restrict__ pTd2, int B) {
  const int b = blockIdx.x * blockDim.x + threadIdx.x;
  if (b >= B) return;
  const float* xr = x + (size_t)b * 18;

  float mn = xr[0], mx = xr[0];
  #pragma unroll
  for (int f = 1; f < 12; ++f) { const float v = xr[f]; mn = fminf(mn, v); mx = fmaxf(mx, v); }
  float rng = ((mx - mn) == 0.0f) ? 1.0f : (mx - mn);
  #pragma unroll
  for (int f = 0; f < 12; ++f) {
    const float p = 0.1f * (xr[f] - mn) / rng;
    pIn[(size_t)b * 12 + f] = fminf(fmaxf(p, 0.0f), 1.0f);
  }

  mn = xr[6]; mx = xr[6];
  #pragma unroll
  for (int f = 7; f < 12; ++f) { const float v = xr[f]; mn = fminf(mn, v); mx = fmaxf(mx, v); }
  rng = ((mx - mn) == 0.0f) ? 1.0f : (mx - mn);
  #pragma unroll
  for (int f = 0; f < 6; ++f) {
    const float p = 0.1f * (xr[6 + f] - mn) / rng;
    pTd[(size_t)b * 6 + f] = fminf(fmaxf(p, 0.0f), 1.0f);
  }

  mn = xr[12]; mx = xr[12];
  #pragma unroll
  for (int f = 13; f < 18; ++f) { const float v = xr[f]; mn = fminf(mn, v); mx = fmaxf(mx, v); }
  rng = ((mx - mn) == 0.0f) ? 1.0f : (mx - mn);
  #pragma unroll
  for (int f = 0; f < 6; ++f) {
    const float p = 0.1f * (xr[12 + f] - mn) / rng;
    pTd2[(size_t)b * 6 + f] = fminf(fmaxf(p, 0.0f), 1.0f);
  }
}

__global__ void snn_head(const float* __restrict__ Wc2, const float* __restrict__ bc2,
                         const float* __restrict__ WA, float* __restrict__ misc) {
  const int t = threadIdx.x;
  if (t < 32) {
    float s = 0.0f;
    for (int ch = 0; ch < 64; ++ch) s += WA[ch] * Wc2[ch * 32 + t];
    misc[t] = s;
  } else if (t == 32) {
    float s = 0.0f;
    for (int ch = 0; ch < 64; ++ch) s += WA[ch] * bc2[ch];
    misc[32] = s;
  }
}

// Pack W1/W2/W3 into split-f16 (hi, lo) B-fragments.
__global__ void snn_pack(const float* __restrict__ W1, const float* __restrict__ W2,
                         const float* __restrict__ W3,
                         unsigned short* __restrict__ w1p,
                         unsigned short* __restrict__ w2p,
                         unsigned short* __restrict__ w3p) {
  const int W1N = 16 * 512;
  const int W2N = 128 * 512;
  const int W3N = 8 * 512;
  const int idx = blockIdx.x * blockDim.x + threadIdx.x;
  if (idx >= W1N + W2N + W3N) return;

  float val = 0.0f;
  unsigned short *dhi, *dlo;
  if (idx < W1N) {
    const int f = idx / 512, pos = idx % 512;
    const int lane = pos / 16, e = pos % 16;
    const int N = f * 16 + (lane % 16);
    const int K = (lane / 16) * 16 + e;
    val = (K < 12) ? W1[(size_t)N * 12 + K] : 0.0f;
    dhi = w1p + idx; dlo = w1p + W1N + idx;
  } else if (idx < W1N + W2N) {
    const int j = idx - W1N;
    const int f = j / 512, pos = j % 512;
    const int kc = f / 16, nt = f % 16;
    const int lane = pos / 16, e = pos % 16;
    const int N = nt * 16 + (lane % 16);
    const int K = kc * 32 + (lane / 16) * 16 + e;
    val = W2[(size_t)N * 256 + K];
    dhi = w2p + j; dlo = w2p + W2N + j;
  } else {
    const int j = idx - W1N - W2N;
    const int kc = j / 512, pos = j % 512;
    const int lane = pos / 16, e = pos % 16;
    const int N = lane % 16;
    const int K = kc * 32 + (lane / 16) * 16 + e;
    val = (N < 6) ? W3[(size_t)N * 256 + K] : 0.0f;
    dhi = w3p + j; dlo = w3p + W3N + j;
  }
  union { _Float16 h; unsigned short u; } cv;
  cv.h = (_Float16)val;
  const float hi_f = (float)cv.h;
  *dhi = cv.u;
  cv.h = (_Float16)(val - hi_f);
  *dlo = cv.u;
}

// ---------------------------------------------------------------------------
// Persistent main kernel: 1 WG = 16 batch rows, 8 waves, T-step loop.
// W2 split-f16 fragments LDS-resident (256KB); W1/W3 fragments in VGPRs;
// m1/m2 persist in WMMA accumulators; noise double-buffered via async-to-LDS.
// Static LDS ~280KB also probes the ISA's 320KB/WG claim.
// ---------------------------------------------------------------------------
__global__ __launch_bounds__(256)
void snn_main(const float* __restrict__ u1, const float* __restrict__ u2,
              const float* __restrict__ u3,
              const float* __restrict__ b1, const float* __restrict__ b2,
              const float* __restrict__ b3,
              const float* __restrict__ Wc1, const float* __restrict__ bc1,
              const float* __restrict__ pIn, const float* __restrict__ pTd,
              const float* __restrict__ pTd2, const float* __restrict__ misc,
              const unsigned short* __restrict__ w1p,
              const unsigned short* __restrict__ w2p,
              const unsigned short* __restrict__ w3p,
              float* __restrict__ out, int B, int T) {
  __shared__ __align__(16) unsigned short ldsW2[2 * 128 * 512]; // hi | lo, 256KB
  __shared__ __align__(16) unsigned short sIn[16 * 40];
  __shared__ __align__(16) unsigned short spk1[16 * 264];
  __shared__ __align__(16) unsigned short spk2[16 * 264];
  __shared__ float cur3[256];
  __shared__ float m3s[16 * 8];
  __shared__ float spk3s[16 * 8];
  __shared__ float sTd[16 * 8];
  __shared__ float sTd2[16 * 8];
  __shared__ __align__(16) float uStage[2][384];  // [buf][u1:0..191 | u2:192..287 | u3:288..383]

  const int tid  = threadIdx.x;
  const int lane = tid & 31;
  const int w    = tid >> 5;
  const int l16  = lane & 15;
  const int B0   = blockIdx.x * 16;

  // ---- one-time: W2 fragments global -> LDS (256KB) ----
#if USE_ASYNC
  {
    const uint4* src = (const uint4*)w2p;
    uint4* dst = (uint4*)ldsW2;
    for (int i = tid; i < 16384; i += 256) async_g2l_b128(src + i, dst + i);
  }
#else
  {
    const uint4* src = (const uint4*)w2p;
    uint4* dst = (uint4*)ldsW2;
    for (int i = tid; i < 16384; i += 256) dst[i] = src[i];
  }
#endif

  // ---- small register-resident fragments: W1, W3 (hi+lo) ----
  v16h w1f[2][2], w3f[2];
  #pragma unroll
  for (int j = 0; j < 2; ++j) {
    w1f[j][0] = load_bfrag(w1p + (2 * w + j) * 512, lane);
    w1f[j][1] = load_bfrag(w1p + 16 * 512 + (2 * w + j) * 512, lane);
  }
  w3f[0] = load_bfrag(w3p + w * 512, lane);
  w3f[1] = load_bfrag(w3p + 8 * 512 + w * 512, lane);

  const float b1v0 = b1[w * 32 + l16];
  const float b1v1 = b1[w * 32 + 16 + l16];
  const float b2v0 = b2[w * 32 + l16];
  const float b2v1 = b2[w * 32 + 16 + l16];
  const float b3j  = b3[tid % 6];

  const int   c4    = lane;
  const float bc1c  = bc1[c4];
  const float wc1a  = Wc1[c4 * 3 + 0];
  const float wc1b  = Wc1[c4 * 3 + 1];
  const float wc1c_ = Wc1[c4 * 3 + 2];
  const float wEffc = misc[c4];
  const float biasA = misc[32];

  // ---- loop-invariant per-thread staging state ----
  const int bI = tid / 12, fI = tid - bI * 12;       // u1 role (tid<192)
  const int b2r = (tid < 96) ? tid / 6 : (tid - 96) / 6;
  const int j2r = (tid < 96) ? tid - b2r * 6 : (tid - 96) - b2r * 6;
  float pInr = 0.0f, pTdr = 0.0f;
  const float* gu1 = u1;
  const float* gu23 = u2;
  if (tid < 192) {
    pInr = pIn[(size_t)(B0 + bI) * 12 + fI];
    gu1  = u1 + (size_t)(B0 + bI) * 12 + fI;
  }
  if (tid < 96) {
    pTdr = pTd[(size_t)(B0 + b2r) * 6 + j2r];
    gu23 = u2 + (size_t)(B0 + b2r) * 6 + j2r;
  } else if (tid < 192) {
    pTdr = pTd2[(size_t)(B0 + b2r) * 6 + j2r];
    gu23 = u3 + (size_t)(B0 + b2r) * 6 + j2r;
  }

  v8f m1a = {}, m1b = {}, m2a = {}, m2b = {};
  float m4[12];
  #pragma unroll
  for (int i = 0; i < 12; ++i) m4[i] = 0.0f;

  sIn[tid] = 0; sIn[tid + 256] = 0;
  if (tid < 128) m3s[tid] = 0.0f;

#if USE_ASYNC
  // waves 0..5 each issue 2 async b32 per staged step (u1 + u2-or-u3)
  auto issue_stage = [&](int tt) {
    const int buf = tt & 1;
    if (tid < 192) async_g2l_b32(gu1 + (size_t)tt * B * 12, &uStage[buf][tid]);
    if (tid < 192) async_g2l_b32(gu23 + (size_t)tt * B * 6, &uStage[buf][192 + tid]);
  };
  __builtin_amdgcn_s_wait_asynccnt(0);   // W2 copy complete
  if (T > 0) issue_stage(0);
#endif
  __syncthreads();

  for (int t = 0; t < T; ++t) {
    // ---- stage input spikes (double-buffered), zero split-K buffer ----
    cur3[tid] = 0.0f;
#if USE_ASYNC
    if (t + 1 < T) {
      issue_stage(t + 1);
      __builtin_amdgcn_s_wait_asynccnt(2);   // step-t transfers done, t+1 in flight
    } else {
      __builtin_amdgcn_s_wait_asynccnt(0);
    }
    const int buf = t & 1;
    if (tid < 192) {
      sIn[bI * 40 + fI] = (uStage[buf][tid] < pInr) ? (unsigned short)0x3C00
                                                    : (unsigned short)0;
      const float uu = uStage[buf][192 + tid];
      if (tid < 96) sTd[b2r * 8 + j2r]  = (uu < pTdr) ? 1.0f : 0.0f;
      else          sTd2[b2r * 8 + j2r] = (uu < pTdr) ? 1.0f : 0.0f;
    }
#else
    if (tid < 192) {
      sIn[bI * 40 + fI] = (gu1[(size_t)t * B * 12] < pInr) ? (unsigned short)0x3C00
                                                           : (unsigned short)0;
      const float uu = gu23[(size_t)t * B * 6];
      if (tid < 96) sTd[b2r * 8 + j2r]  = (uu < pTdr) ? 1.0f : 0.0f;
      else          sTd2[b2r * 8 + j2r] = (uu < pTdr) ? 1.0f : 0.0f;
      if (t + 1 < T) {
        __builtin_prefetch(gu1 + (size_t)(t + 1) * B * 12, 0, 1);
        __builtin_prefetch(gu23 + (size_t)(t + 1) * B * 6, 0, 1);
      }
    }
#endif
    __syncthreads();

    // ---- layer 1: 16x12 @ 12x256 (split-f16) ----
    {
      v16h a = load_afrag(sIn, 40, 0, lane);
      v8f z = {};
      v8f c0 = __builtin_amdgcn_wmma_f32_16x16x32_f16(false, a, false, w1f[0][0], (short)0, z, false, false);
      c0     = __builtin_amdgcn_wmma_f32_16x16x32_f16(false, a, false, w1f[0][1], (short)0, c0, false, false);
      v8f c1 = __builtin_amdgcn_wmma_f32_16x16x32_f16(false, a, false, w1f[1][0], (short)0, z, false, false);
      c1     = __builtin_amdgcn_wmma_f32_16x16x32_f16(false, a, false, w1f[1][1], (short)0, c1, false, false);
      lif_update_store(c0, m1a, b1v0, spk1, 264, w * 32, lane);
      lif_update_store(c1, m1b, b1v1, spk1, 264, w * 32 + 16, lane);
    }
    __syncthreads();

    // ---- layer 2: 16x256 @ 256x256, LDS-resident split-f16 B-fragments ----
    {
      v8f acc0 = {}, acc1 = {};
      #pragma unroll
      for (int kc = 0; kc < 8; ++kc) {
        v16h a = load_afrag(spk1, 264, kc * 32, lane);
        const unsigned short* f0 = ldsW2 + (kc * 16 + 2 * w + 0) * 512;
        const unsigned short* f1 = ldsW2 + (kc * 16 + 2 * w + 1) * 512;
        acc0 = __builtin_amdgcn_wmma_f32_16x16x32_f16(false, a, false, load_bfrag(f0, lane), (short)0, acc0, false, false);
        acc0 = __builtin_amdgcn_wmma_f32_16x16x32_f16(false, a, false, load_bfrag(f0 + 128 * 512, lane), (short)0, acc0, false, false);
        acc1 = __builtin_amdgcn_wmma_f32_16x16x32_f16(false, a, false, load_bfrag(f1, lane), (short)0, acc1, false, false);
        acc1 = __builtin_amdgcn_wmma_f32_16x16x32_f16(false, a, false, load_bfrag(f1 + 128 * 512, lane), (short)0, acc1, false, false);
      }
      lif_update_store(acc0, m2a, b2v0, spk2, 264, w * 32, lane);
      lif_update_store(acc1, m2b, b2v1, spk2, 264, w * 32 + 16, lane);
    }
    __syncthreads();

    // ---- layer 3: split-K over 8 waves (split-f16), LDS ds_add_f32 reduce ----
    {
      v16h a = load_afrag(spk2, 264, w * 32, lane);
      v8f z = {};
      v8f part = __builtin_amdgcn_wmma_f32_16x16x32_f16(false, a, false, w3f[0], (short)0, z, false, false);
      part     = __builtin_amdgcn_wmma_f32_16x16x32_f16(false, a, false, w3f[1], (short)0, part, false, false);
      const int hi = (lane >> 4) & 1;
      #pragma unroll
      for (int r = 0; r < 8; ++r) {
        const int M = r + (hi << 3);
        atomicAdd(&cur3[M * 16 + l16], part[r]);
      }
    }
    __syncthreads();

    if (tid < 96) {
      const int b = tid / 6, j = tid - b * 6;
      const float c = cur3[b * 16 + j] + b3j;
      float mm = m3s[b * 8 + j];
      const float reset = (mm > THR) ? THR : 0.0f;
      mm = BETA * mm + c - reset;
      m3s[b * 8 + j]   = mm;
      spk3s[b * 8 + j] = (mm > THR) ? 1.0f : 0.0f;
    }
    __syncthreads();

    // ---- layer 4: 1x1 conv (3->32) + LIF; fused conv2+head at t == T-1 ----
    #pragma unroll
    for (int i = 0; i < 12; ++i) {
      const int bp  = w * 12 + i;
      const int b   = bp / 6;
      const int pos = bp - b * 6;
      const float cur = bc1c + wc1a * sTd[b * 8 + pos]
                             + wc1b * spk3s[b * 8 + pos]
                             + wc1c_ * sTd2[b * 8 + pos];
      float mm = m4[i];
      const float reset = (mm > THR) ? THR : 0.0f;
      mm = BETA * mm + cur - reset;
      m4[i] = mm;
      if (t == T - 1) {
        float v = ((mm > THR) ? 1.0f : 0.0f) * wEffc;
        #pragma unroll
        for (int off = 16; off > 0; off >>= 1) v += __shfl_xor(v, off, 32);
        if (lane == 0 && pos < 3) out[(size_t)(B0 + b) * 3 + pos] = biasA + v;
      }
    }
    __syncthreads();
  }
}

// ---------------------------------------------------------------------------
extern "C" void kernel_launch(void* const* d_in, const int* in_sizes, int n_in,
                              void* d_out, int out_size, void* d_ws, size_t ws_size,
                              hipStream_t stream) {
  const float* x   = (const float*)d_in[0];
  const float* u1  = (const float*)d_in[1];
  const float* u2  = (const float*)d_in[2];
  const float* u3  = (const float*)d_in[3];
  const float* W1  = (const float*)d_in[4];
  const float* b1  = (const float*)d_in[5];
  const float* W2  = (const float*)d_in[6];
  const float* b2  = (const float*)d_in[7];
  const float* W3  = (const float*)d_in[8];
  const float* b3  = (const float*)d_in[9];
  const float* Wc1 = (const float*)d_in[10];
  const float* bc1 = (const float*)d_in[11];
  const float* Wc2 = (const float*)d_in[12];
  const float* bc2 = (const float*)d_in[13];
  const float* WA  = (const float*)d_in[14];
  float* out = (float*)d_out;

  const int B = in_sizes[0] / 18;
  const int T = in_sizes[1] / (B * 12);

  char* ws = (char*)d_ws;
  size_t off = 0;
  float* pIn  = (float*)(ws + off); off += (size_t)B * 12 * 4;
  float* pTd  = (float*)(ws + off); off += (size_t)B * 6 * 4;
  float* pTd2 = (float*)(ws + off); off += (size_t)B * 6 * 4;
  float* misc = (float*)(ws + off); off += 64 * 4;
  unsigned short* w1p = (unsigned short*)(ws + off); off += (size_t)2 * 16 * 512 * 2;
  unsigned short* w2p = (unsigned short*)(ws + off); off += (size_t)2 * 128 * 512 * 2;
  unsigned short* w3p = (unsigned short*)(ws + off); off += (size_t)2 * 8 * 512 * 2;

  snn_probs<<<(B + 255) / 256, 256, 0, stream>>>(x, pIn, pTd, pTd2, B);
  snn_head<<<1, 64, 0, stream>>>(Wc2, bc2, WA, misc);
  const int packN = (16 + 128 + 8) * 512;
  snn_pack<<<(packN + 255) / 256, 256, 0, stream>>>(W1, W2, W3, w1p, w2p, w3p);
  snn_main<<<B / 16, 256, 0, stream>>>(u1, u2, u3, b1, b2, b3, Wc1, bc1,
                                       pIn, pTd, pTd2, misc, w1p, w2p, w3p,
                                       out, B, T);
}
